// EdgeConv_Mask_45174466019828
// MI455X (gfx1250) — matrix operand order repaired and checked by reading, not compiled, not run on hardware
//
#include <hip/hip_runtime.h>
#include <hip/hip_bf16.h>

typedef __attribute__((ext_vector_type(16))) __bf16   v16bf;
typedef __attribute__((ext_vector_type(8)))  __bf16   v8bf;
typedef __attribute__((ext_vector_type(8)))  float    v8f;
typedef __attribute__((ext_vector_type(4)))  float    f32x4;
typedef __attribute__((ext_vector_type(4)))  unsigned u32x4;

#define DIM       64
#define W1_STRIDE 100   // u32 words per output column (96 used + 4 pad; %4==0 keeps 16B align, breaks 64-bank conflicts)
#define W2_STRIDE 36    // u32 words per output column (32 used + 4 pad)
#define H_STRIDE  72    // bf16 per staged h row (64 used + 8 pad; 144B keeps 16B alignment)

__device__ __forceinline__ unsigned pack_bf16x2(float lo, float hi) {
    __bf16 a = (__bf16)lo, b = (__bf16)hi;
    unsigned short ua, ub;
    __builtin_memcpy(&ua, &a, 2);
    __builtin_memcpy(&ub, &b, 2);
    return ((unsigned)ub << 16) | (unsigned)ua;
}

__global__ __launch_bounds__(256, 1)
void edgeconv_wmma_kernel(const float* __restrict__ x,
                          const float* __restrict__ ea,
                          const float* __restrict__ W1,
                          const float* __restrict__ b1,
                          const float* __restrict__ W2,
                          const float* __restrict__ b2,
                          const int*   __restrict__ eidx,
                          float*       __restrict__ out,
                          int nE, int nV)
{
    // Weights pre-packed into the bf16 B-operand register layout:
    //   word[col*STRIDE + k2] = { W[2*k2+1][col] , W[2*k2][col] }
    __shared__ __align__(16) unsigned sW1[64 * W1_STRIDE];   // 25.6 KB
    __shared__ __align__(16) unsigned sW2[64 * W2_STRIDE];   //  9.2 KB
    __shared__ __align__(16) __bf16   sH[8 * 16 * H_STRIDE]; // 18.4 KB, wave-private, reused per M-tile

    const int tid = threadIdx.x;

    // ---- Stage W1 (192x64) and W2 (64x64) as packed bf16 K-pairs ----
    for (int p = tid; p < 96 * 64; p += 256) {
        const int n = p & 63, k2 = p >> 6;
        sW1[n * W1_STRIDE + k2] = pack_bf16x2(W1[(2 * k2) * DIM + n],
                                              W1[(2 * k2 + 1) * DIM + n]);
    }
    for (int p = tid; p < 32 * 64; p += 256) {
        const int n = p & 63, k2 = p >> 6;
        sW2[n * W2_STRIDE + k2] = pack_bf16x2(W2[(2 * k2) * DIM + n],
                                              W2[(2 * k2 + 1) * DIM + n]);
    }
    __syncthreads();

    const int wave   = tid >> 5;
    const int lane   = tid & 31;
    const int ln     = lane & 15;
    const int hi     = lane >> 4;              // upper half-wave flag
    const int base32 = (blockIdx.x * 8 + wave) * 32;   // 32 edges (2 M-tiles) per wave
    const int base0  = hi ? 8 : 0;             // ISA 16-bit A/B lane-half K offset

    // ---- Gather bases for both A M-tiles (row M = ln within each tile) ----
    const float* srcs[2][3];
#pragma unroll
    for (int m = 0; m < 2; ++m) {
        int eA = base32 + m * 16 + ln;
        if (eA > nE - 1) eA = nE - 1;          // clamp (keeps EXEC uniform for WMMA)
        int r = eidx[eA];
        int c = eidx[nE + eA];
        if ((unsigned)r >= (unsigned)nV) r = 0;
        if ((unsigned)c >= (unsigned)nV) c = 0;
        srcs[m][0] = x  + (size_t)r  * DIM;    // K   0..63 : x[row]   (L2-resident, RT)
        srcs[m][1] = x  + (size_t)c  * DIM;    // K  64..127: x[col]   (L2-resident, RT)
        srcs[m][2] = ea + (size_t)eA * DIM;    // K 128..191: edge_attr (streamed once, NT)
    }

    float b1v[4], b2v[4];
#pragma unroll
    for (int t = 0; t < 4; ++t) {
        b1v[t] = b1[t * 16 + ln];
        b2v[t] = b2[t * 16 + ln];
    }

    // ========== Layer 1: h = relu([x_r|x_c|ea] @ W1 + b1), 2 M-tiles share B ==========
    v8f acc[2][4] = {};                        // [M-tile][N-tile], 16x16 f32 each
#pragma unroll
    for (int ck = 0; ck < 6; ++ck) {           // K chunks of 32
        v16bf a[2];
#pragma unroll
        for (int m = 0; m < 2; ++m) {
            const float* p = srcs[m][ck >> 1] + (ck & 1) * 32 + base0;
            f32x4 f0, f1, f2, f3;
            if (ck >= 4) {                     // edge_attr: streaming, keep out of L2 working set
                f0 = __builtin_nontemporal_load((const f32x4*)(p));
                f1 = __builtin_nontemporal_load((const f32x4*)(p + 4));
                f2 = __builtin_nontemporal_load((const f32x4*)(p + 16));
                f3 = __builtin_nontemporal_load((const f32x4*)(p + 20));
            } else {                           // x gathers: want L2 residency (12.8 MB << 192 MB)
                f0 = *(const f32x4*)(p);
                f1 = *(const f32x4*)(p + 4);
                f2 = *(const f32x4*)(p + 16);
                f3 = *(const f32x4*)(p + 20);
            }
#pragma unroll
            for (int i = 0; i < 4; ++i) {
                a[m][i]      = (__bf16)f0[i];
                a[m][4 + i]  = (__bf16)f1[i];
                a[m][8 + i]  = (__bf16)f2[i];
                a[m][12 + i] = (__bf16)f3[i];
            }
        }
#pragma unroll
        for (int t = 0; t < 4; ++t) {
            const int col = t * 16 + ln;
            const unsigned* wp = &sW1[col * W1_STRIDE + ck * 16 + hi * 4];
            union { u32x4 q[2]; v16bf v; } B;
            B.q[0] = *(const u32x4*)(wp);
            B.q[1] = *(const u32x4*)(wp + 8);
#pragma unroll
            for (int m = 0; m < 2; ++m)        // B reused for both M-tiles
                acc[m][t] = __builtin_amdgcn_wmma_f32_16x16x32_bf16(
                                false, a[m], false, B.v, (short)0, acc[m][t], false, false);
        }
    }

    // ========== Per M-tile: bias+ReLU -> LDS transpose -> Layer 2 -> store ==========
    __bf16* htile = &sH[(size_t)wave * 16 * H_STRIDE];
    const __bf16* myrow = &htile[ln * H_STRIDE];
#pragma unroll
    for (int m = 0; m < 2; ++m) {
        // C layout (VGPR rr -> row, lane -> col) -> row-major bf16 in wave-private LDS
#pragma unroll
        for (int t = 0; t < 4; ++t) {
#pragma unroll
            for (int rr = 0; rr < 8; ++rr) {
                float v = acc[m][t][rr] + b1v[t];
                v = v > 0.f ? v : 0.f;
                htile[(hi * 8 + rr) * H_STRIDE + t * 16 + ln] = (__bf16)v;
            }
        }
        // wave-local LDS RAW/WAR: DS ops are in-order per wave; compiler inserts s_wait_dscnt.

        v8f acc2[4] = {};
#pragma unroll
        for (int ck = 0; ck < 2; ++ck) {       // K chunks of 32
            const __bf16* ap = myrow + ck * 32 + base0;
            union { v8bf h[2]; v16bf v; } A;
            A.h[0] = *(const v8bf*)(ap);
            A.h[1] = *(const v8bf*)(ap + 16);
#pragma unroll
            for (int t = 0; t < 4; ++t) {
                const int col = t * 16 + ln;
                const unsigned* wp = &sW2[col * W2_STRIDE + ck * 16 + hi * 4];
                union { u32x4 q[2]; v16bf v; } B;
                B.q[0] = *(const u32x4*)(wp);
                B.q[1] = *(const u32x4*)(wp + 8);
                acc2[t] = __builtin_amdgcn_wmma_f32_16x16x32_bf16(
                              false, A.v, false, B.v, (short)0, acc2[t], false, false);
            }
        }

        // Output is written once and never re-read: non-temporal stores keep it out of L2.
#pragma unroll
        for (int t = 0; t < 4; ++t) {
#pragma unroll
            for (int rr = 0; rr < 8; ++rr) {
                const int eOut = base32 + m * 16 + hi * 8 + rr;
                if (eOut < nE)
                    __builtin_nontemporal_store(acc2[t][rr] + b2v[t],
                                                out + (size_t)eOut * DIM + t * 16 + ln);
            }
        }
    }
}

extern "C" void kernel_launch(void* const* d_in, const int* in_sizes, int n_in,
                              void* d_out, int out_size, void* d_ws, size_t ws_size,
                              hipStream_t stream) {
    const float* x  = (const float*)d_in[0];
    const float* ea = (const float*)d_in[1];
    const float* W1 = (const float*)d_in[2];
    const float* b1 = (const float*)d_in[3];
    const float* W2 = (const float*)d_in[4];
    const float* b2 = (const float*)d_in[5];
    const int*   ei = (const int*)d_in[6];
    // d_in[7] = edge_type (unused by the reference computation)
    float* out = (float*)d_out;

    const int nV = in_sizes[0] / DIM;
    const int nE = in_sizes[1] / DIM;
    const int blocks = (nE + 255) / 256;       // 256 edges per 256-thread block (32 per wave)

    hipLaunchKernelGGL(edgeconv_wmma_kernel, dim3(blocks), dim3(256), 0, stream,
                       x, ea, W1, b1, W2, b2, ei, out, nE, nV);
}
